// MargKernel_50955492000052
// MI455X (gfx1250) — compile-verified, matrix operand order checked by reading, not run on hardware
//
#include <hip/hip_runtime.h>
#include <math.h>

#define N_PTS  4096
#define KMODES 128
#define DDIM   128
#define TN     128           // rows of x per workgroup
#define BST    136           // bf16 LDS row stride (272B = 68 dwords -> rotating banks)
#define QP_ST  17            // padded stride for partial-q array

typedef __attribute__((ext_vector_type(16))) __bf16 v16bf;
typedef __attribute__((ext_vector_type(8)))  __bf16 bf16x8;
typedef __attribute__((ext_vector_type(4)))  __bf16 bf16x4;
typedef __attribute__((ext_vector_type(8)))  float  v8f;

// ---------------------------------------------------------------------------
// Kernel 0: per-mode constants c[k] = logdet_k + log_softmax(weigh)_k ; zero out
// ---------------------------------------------------------------------------
__global__ __launch_bounds__(128)
void prep_kernel(const float* __restrict__ logvar,
                 const float* __restrict__ weigh,
                 float* __restrict__ c,
                 float* __restrict__ out) {
    __shared__ float red[KMODES];
    int k = threadIdx.x;
    float wk = weigh[k];

    red[k] = wk; __syncthreads();
    for (int off = 64; off > 0; off >>= 1) {
        if (k < off) red[k] = fmaxf(red[k], red[k + off]);
        __syncthreads();
    }
    float mx = red[0]; __syncthreads();

    red[k] = expf(wk - mx); __syncthreads();
    for (int off = 64; off > 0; off >>= 1) {
        if (k < off) red[k] += red[k + off];
        __syncthreads();
    }
    float lsew = mx + logf(red[0]);

    const float* lvk = logvar + (size_t)k * DDIM;
    float logdet = 0.f;
    for (int d = 0; d < DDIM; ++d) {
        float v = expf(tanhf(lvk[d]));
        logdet += logf(fabsf(v) + 1e-8f);
    }
    c[k] = logdet + (wk - lsew);
    if (k == 0) out[0] = 0.f;
}

// split a float into hi/lo bf16 (hi = RNE(x), lo = RNE(x - hi))
__device__ __forceinline__ __bf16 bf_hi(float x) { return (__bf16)x; }
__device__ __forceinline__ __bf16 bf_lo(float x) {
    __bf16 h = (__bf16)x;
    return (__bf16)(x - (float)h);
}

// load a 16x32 bf16 fragment chunk: lane(half,ln) takes K = [k0+8h, +7] and [k0+16+8h, +7]
__device__ __forceinline__ v16bf load_frag(const __bf16* rowp, int kb) {
    bf16x8 c0 = *(const bf16x8*)(rowp + kb * 32);
    bf16x8 c1 = *(const bf16x8*)(rowp + kb * 32 + 16);
    return __builtin_shufflevector(c0, c1, 0, 1, 2, 3, 4, 5, 6, 7,
                                           8, 9, 10, 11, 12, 13, 14, 15);
}

// ---------------------------------------------------------------------------
// Kernel 1: per (n-tile, mode):
//   Yin = (x-mu)*var ; Yfinal = Yin @ (L+I)^T via split-bf16 WMMA
//   (3x V_WMMA_F32_16X16X32_BF16 per 32-K chunk: ah*bh + ah*bl + al*bh,
//    fp32-class accuracy) ; q = ||Yfinal||^2 ; st[k*N+n] = -q/2 + c[k]
// ---------------------------------------------------------------------------
__global__ __launch_bounds__(256)
void gmm_main_kernel(const float* __restrict__ x,
                     const float* __restrict__ means,
                     const float* __restrict__ logvar,
                     const float* __restrict__ tri,
                     const float* __restrict__ c,
                     float* __restrict__ st) {
    __shared__ __bf16 Ah[TN * BST];     // Yin hi, row-major [n][d]
    __shared__ __bf16 Al[TN * BST];     // Yin lo
    __shared__ __bf16 Bh[DDIM * BST];   // (L+I) hi, row-major [i][j]
    __shared__ __bf16 Bl[DDIM * BST];   // (L+I) lo
    __shared__ float  ms[DDIM];
    __shared__ float  vs[DDIM];
    __shared__ float  qp[TN * QP_ST];   // per-(row, lane-col) partial squares
    __shared__ float  ck;

    const int kmode = blockIdx.y;
    const int n0    = blockIdx.x * TN;
    const int tid   = threadIdx.x;

    // stage 0: per-mode mean / var, zero partial-q
    if (tid < DDIM) {
        vs[tid] = expf(tanhf(logvar[(size_t)kmode * DDIM + tid]));
        ms[tid] = means[(size_t)kmode * DDIM + tid];
    }
    for (int e = tid; e < TN * QP_ST; e += 256) qp[e] = 0.f;
    if (tid == 0) ck = c[kmode];
    __syncthreads();

    // stage 1: Yin = (x - mu) * var -> split into Ah/Al (float4 global loads)
    for (int e4 = tid; e4 < TN * DDIM / 4; e4 += 256) {
        int r = e4 >> 5;
        int d = (e4 & 31) * 4;
        float4 xv = ((const float4*)x)[(size_t)(n0 + r) * (DDIM / 4) + (e4 & 31)];
        float y0 = (xv.x - ms[d + 0]) * vs[d + 0];
        float y1 = (xv.y - ms[d + 1]) * vs[d + 1];
        float y2 = (xv.z - ms[d + 2]) * vs[d + 2];
        float y3 = (xv.w - ms[d + 3]) * vs[d + 3];
        bf16x4 h = { bf_hi(y0), bf_hi(y1), bf_hi(y2), bf_hi(y3) };
        bf16x4 l = { bf_lo(y0), bf_lo(y1), bf_lo(y2), bf_lo(y3) };
        *(bf16x4*)&Ah[r * BST + d] = h;
        *(bf16x4*)&Al[r * BST + d] = l;
    }
    // stage 2: Bs[i][j] = strict-lower tri + identity on diagonal -> Bh/Bl
    const float* trik = tri + (size_t)kmode * DDIM * DDIM;
    for (int e4 = tid; e4 < DDIM * DDIM / 4; e4 += 256) {
        int i = e4 >> 5;
        int j = (e4 & 31) * 4;
        float4 t = ((const float4*)trik)[e4];
        float b0 = (i > j + 0) ? t.x : (i == j + 0 ? 1.f : 0.f);
        float b1 = (i > j + 1) ? t.y : (i == j + 1 ? 1.f : 0.f);
        float b2 = (i > j + 2) ? t.z : (i == j + 2 ? 1.f : 0.f);
        float b3 = (i > j + 3) ? t.w : (i == j + 3 ? 1.f : 0.f);
        bf16x4 h = { bf_hi(b0), bf_hi(b1), bf_hi(b2), bf_hi(b3) };
        bf16x4 l = { bf_lo(b0), bf_lo(b1), bf_lo(b2), bf_lo(b3) };
        *(bf16x4*)&Bh[i * BST + j] = h;
        *(bf16x4*)&Bl[i * BST + j] = l;
    }
    __syncthreads();

    const int wave = __builtin_amdgcn_readfirstlane(tid) >> 5;  // scalar wave id
    const int lane = tid & 31;
    const int half = lane >> 4;
    const int ln   = lane & 15;   // M for A-frag, N for B-frag

    for (int tm = 0; tm < 8; ++tm) {
        // balanced pairing: every wave runs exactly 20 K-chunks (60 WMMAs)
        const int tn = (tm < 4) ? wave : 7 - wave;
        v8f acc = {};

        const __bf16* ahp = &Ah[(tm * 16 + ln) * BST + 8 * half];
        const __bf16* alp = &Al[(tm * 16 + ln) * BST + 8 * half];
        const __bf16* bhp = &Bh[(tn * 16 + ln) * BST + 8 * half];
        const __bf16* blp = &Bl[(tn * 16 + ln) * BST + 8 * half];

        // K chunks of 32 covering j <= tn*16+15 (upper part of B is zero-masked,
        // so the rounded-up half chunk is harmless)
        const int nk32 = (tn >> 1) + 1;

        for (int kb = 0; kb < nk32; ++kb) {
            v16bf ah = load_frag(ahp, kb);
            v16bf al = load_frag(alp, kb);
            v16bf bh = load_frag(bhp, kb);
            v16bf bl = load_frag(blp, kb);
            acc = __builtin_amdgcn_wmma_f32_16x16x32_bf16(false, ah, false, bh,
                                                          (short)0, acc, false, false);
            acc = __builtin_amdgcn_wmma_f32_16x16x32_bf16(false, ah, false, bl,
                                                          (short)0, acc, false, false);
            acc = __builtin_amdgcn_wmma_f32_16x16x32_bf16(false, al, false, bh,
                                                          (short)0, acc, false, false);
        }

        // epilogue: acc == Yfinal tile; fire-and-forget LDS atomic partials
        const int base = (tm * 16 + 8 * half) * QP_ST + ln;
        #pragma unroll
        for (int v = 0; v < 8; ++v) {
            float fv = acc[v];
            atomicAdd(&qp[base + v * QP_ST], fv * fv);   // ds_add_f32, no return
        }
    }
    __syncthreads();

    if (tid < TN) {
        float q = 0.f;
        #pragma unroll
        for (int c2 = 0; c2 < 16; ++c2) q += qp[tid * QP_ST + c2];
        st[(size_t)kmode * N_PTS + n0 + tid] = -0.5f * q + ck;
    }
}

// ---------------------------------------------------------------------------
// Kernel 2: per point logsumexp over K, mean reduce into out[0]
// ---------------------------------------------------------------------------
__global__ __launch_bounds__(256)
void lse_kernel(const float* __restrict__ st, float* __restrict__ out) {
    __shared__ float red[256];
    const int n = blockIdx.x * 256 + threadIdx.x;

    float m = -INFINITY, s = 0.f;
    for (int k = 0; k < KMODES; ++k) {
        float v = st[(size_t)k * N_PTS + n];   // coalesced across threads
        if (v > m) { s = s * expf(m - v) + 1.f; m = v; }
        else       { s += expf(v - m); }
    }
    float lse = m + logf(s);
    if (isinf(lse) && lse < 0.f) lse = -100000.f;

    const float logC = -117.62439325563593f;   // -(D/2)*log(2*pi), D=128
    float val = -(logC + lse) * (1.0f / (float)N_PTS);

    red[threadIdx.x] = val; __syncthreads();
    for (int off = 128; off > 0; off >>= 1) {
        if (threadIdx.x < off) red[threadIdx.x] += red[threadIdx.x + off];
        __syncthreads();
    }
    if (threadIdx.x == 0) atomicAdd(out, red[0]);
}

// ---------------------------------------------------------------------------
extern "C" void kernel_launch(void* const* d_in, const int* in_sizes, int n_in,
                              void* d_out, int out_size, void* d_ws, size_t ws_size,
                              hipStream_t stream) {
    const float* x      = (const float*)d_in[0];
    const float* means  = (const float*)d_in[1];
    const float* logvar = (const float*)d_in[2];
    const float* tri    = (const float*)d_in[3];
    const float* weigh  = (const float*)d_in[4];
    float* out = (float*)d_out;

    float* c  = (float*)d_ws;          // K floats
    float* st = c + 256;               // K*N floats (transposed s)

    prep_kernel<<<1, 128, 0, stream>>>(logvar, weigh, c, out);

    dim3 grid(N_PTS / TN, KMODES);
    gmm_main_kernel<<<grid, 256, 0, stream>>>(x, means, logvar, tri, c, st);

    lse_kernel<<<N_PTS / 256, 256, 0, stream>>>(st, out);
}
